// SampleNet_69612829933913
// MI455X (gfx1250) — matrix operand order repaired and checked by reading, not compiled, not run on hardware
//
#include <hip/hip_runtime.h>
#include <hip/hip_bf16.h>
#include <stdint.h>

typedef __attribute__((ext_vector_type(16))) _Float16 v16h;
typedef __attribute__((ext_vector_type(8)))  _Float16 v8h;
typedef __attribute__((ext_vector_type(8)))  float    v8f;

#define HW 224
#define NPIX (HW*HW)            // 50176
#define BATCH 8
#define LLEN 64
#define NFR (BATCH*LLEN)        // 512
#define RPB 32                  // output rows per block (strip height)
#define RBLOCKS (HW/RPB)        // 7 strips per frame
#define SROWS_USED 38           // 32 + 6 halo rows
#define SROWS_ALLOC 48          // padded so A1 v8h loads stay in-bounds (zeroed)
#define SSTRIDE 240             // f16 row stride (mult of 8 -> 16B-aligned chunks)
#define SCOLS 230               // 3 + 224 + 3 halo cols
#define NSUB 28                 // 2 x 14 16x16 subtiles per strip
#define HSTRIDE (32*16)         // h panel padded to 32 rows: unconditional stores

// ------------------------------------------------------------------
// Kernel 1: conv(3->1, 7x7, all-equal weights) == scale * box7x7(sum_c x).
// Workgroup loads one 32x224 strip's channel-sum halo (f16) into LDS once
// (halo amplification 1.22x). Each wave loops over 16x16 subtiles:
// horizontal 7-tap box sum as V_WMMA_F32_16X16X32_F16 against a
// register-built banded 0/1 matrix (K=32 covers the 16+6 column halo
// exactly), vertical 7-tap via VALU. h panels padded to 32 rows so all
// C-fragment scatters are unconditional (no exec-mask divergence).
// ------------------------------------------------------------------
__global__ __launch_bounds__(256) void conv_boxsum_wmma(
    const float* __restrict__ x, const float* __restrict__ cw,
    const float* __restrict__ cb, float* __restrict__ y)
{
  __shared__ __attribute__((aligned(16))) _Float16 sT[SROWS_ALLOC * SSTRIDE];
  __shared__ float hT[NSUB * HSTRIDE];

  const int tid  = threadIdx.x;
  const int wv   = tid >> 5;
  const int lane = tid & 31;

  const int frame = blockIdx.x / RBLOCKS;
  const int tr    = (blockIdx.x % RBLOCKS) * RPB;

  // zero-fill sT (halo + pad rows/cols must be 0 and finite for WMMA)
  {
    uint32_t* zp = (uint32_t*)sT;
    for (int i = tid; i < SROWS_ALLOC * SSTRIDE / 2; i += 256) zp[i] = 0u;
  }
  __syncthreads();

  // cooperative load of channel sums: image rows tr-3..tr+34, cols -3..226
  const float* xf = x + (size_t)frame * (3 * NPIX);
  for (int e = tid; e < SROWS_USED * SCOLS; e += 256) {
    int rr = e / SCOLS, cc = e - rr * SCOLS;
    int gr = tr - 3 + rr, gc = cc - 3;
    if ((unsigned)gr < HW && (unsigned)gc < HW) {
      int p = gr * HW + gc;
      float s = xf[p] + xf[p + NPIX] + xf[p + 2 * NPIX];
      sT[rr * SSTRIDE + cc] = (_Float16)s;
    }
  }
  __syncthreads();

  // Band matrix B (32x16), built once per wave: lane&15 = N, element e = K (table order);
  // B[k][n] = (n <= k <= n+6)
  v16h bb;
  #pragma unroll
  for (int e = 0; e < 16; ++e) {
    const int bk = (e < 8) ? ((lane >> 4) * 8 + e) : (16 + (lane >> 4) * 8 + (e - 8));
    const int bn = lane & 15;
    bb[e] = (bk >= bn && bk <= bn + 6) ? (_Float16)1.0f : (_Float16)0.0f;
  }

  const int hi = lane >> 4;
  const int M  = lane & 15;

  // each wave handles subtiles st, st+8, ... (no barriers inside loop)
  for (int st = wv; st < NSUB; st += 8) {
    const int str = st / 14, stc = st - str * 14;
    const int R    = str * 16;          // subtile row base within strip
    const int cofs = stc * 16 + hi * 8; // f16 A-layout: two runs of 8 K's

    const _Float16* b0 = &sT[(R + M) * SSTRIDE + cofs];
    const _Float16* b1 = &sT[(R + 16 + M) * SSTRIDE + cofs];
    v8h a0lo = *(const v8h*)(b0);
    v8h a0hi = *(const v8h*)(b0 + 16);
    v8h a1lo = *(const v8h*)(b1);
    v8h a1hi = *(const v8h*)(b1 + 16);
    v16h a0, a1;
    #pragma unroll
    for (int i = 0; i < 8; ++i) {
      a0[i] = a0lo[i]; a0[i + 8] = a0hi[i];
      a1[i] = a1lo[i]; a1[i + 8] = a1hi[i];
    }

    v8f c0 = {}; v8f c1 = {};
    c0 = __builtin_amdgcn_wmma_f32_16x16x32_f16(false, a0, false, bb, (short)0, c0, false, false);
    c1 = __builtin_amdgcn_wmma_f32_16x16x32_f16(false, a1, false, bb, (short)0, c1, false, false);

    // scatter horizontal sums into this subtile's padded 32x16 h panel
    // (rows 22..31 are dead padding -> every store unconditional)
    float* hp = &hT[st * HSTRIDE + hi * (8 * 16) + M];
    #pragma unroll
    for (int v = 0; v < 8; ++v) {
      hp[v * 16]            = c0[v];     // h rows 0..15
      hp[v * 16 + 16 * 16]  = c1[v];     // h rows 16..31 (16..21 used)
    }
  }
  __syncthreads();

  // vertical 7-tap sum, scale by conv weight (all taps equal), add bias
  const float scale = cw[0];
  const float bias  = cb[0];
  float* yf = y + (size_t)frame * NPIX + (size_t)tr * HW;
  for (int i = tid; i < RPB * HW; i += 256) {
    int r = i / HW, c = i - r * HW;
    int st = (r >> 4) * 14 + (c >> 4);
    const float* hp = &hT[st * HSTRIDE + (r & 15) * 16 + (c & 15)];
    float acc = 0.f;
    #pragma unroll
    for (int k = 0; k < 7; ++k) acc += hp[k * 16];
    yf[i] = acc * scale + bias;          // coalesced (consecutive c per lane)
  }
}

// ------------------------------------------------------------------
// Kernel 2: diff_score[b,l] = sum_p |y[b,l,p] - y[b,l+1,p] + eps|
// float4-vectorized stream + prefetch (global_prefetch_b8).
// ------------------------------------------------------------------
__global__ __launch_bounds__(256) void diff_kernel(
    const float* __restrict__ y, float* __restrict__ diff)
{
  __shared__ float red[256];
  const int b = blockIdx.x / (LLEN - 1);
  const int l = blockIdx.x % (LLEN - 1);
  const float4* ya = (const float4*)(y + ((size_t)(b * LLEN + l)) * NPIX);
  const float4* yb = (const float4*)(y + ((size_t)(b * LLEN + l + 1)) * NPIX);
  float acc = 0.f;
  for (int p = threadIdx.x; p < NPIX / 4; p += 256) {
    __builtin_prefetch(&ya[p + 512], 0, 1);
    __builtin_prefetch(&yb[p + 512], 0, 1);
    float4 a = ya[p], bv = yb[p];
    acc += fabsf(a.x - bv.x + 1e-6f) + fabsf(a.y - bv.y + 1e-6f)
         + fabsf(a.z - bv.z + 1e-6f) + fabsf(a.w - bv.w + 1e-6f);
  }
  red[threadIdx.x] = acc;
  __syncthreads();
  for (int s = 128; s > 0; s >>= 1) {
    if (threadIdx.x < s) red[threadIdx.x] += red[threadIdx.x + s];
    __syncthreads();
  }
  if (threadIdx.x == 0) diff[blockIdx.x] = red[0];
}

// ------------------------------------------------------------------
// Kernel 3: dsp = diff^0.5 (ratio = (16/64)^0.5 = 0.5), normalized cumsum,
// first-argmin against 16 uniform sample points -> frame indices.
// ------------------------------------------------------------------
__global__ void select_kernel(const float* __restrict__ diff, int* __restrict__ idx)
{
  const int b = threadIdx.x;
  if (b >= BATCH) return;
  float cum[LLEN - 1];
  float s = 0.f;
  for (int l = 0; l < LLEN - 1; ++l) {
    float d = sqrtf(diff[b * (LLEN - 1) + l]);
    s += d;
    cum[l] = s;
  }
  const float inv = 1.f / s;
  for (int m = 0; m < 16; ++m) {
    float target = (float)m * (1.f / 15.f);
    float best = 3.4e38f;
    int bi = 0;
    for (int l = 0; l < LLEN - 1; ++l) {
      float v = fabsf(cum[l] * inv - target);
      if (v < best) { best = v; bi = l; }   // first occurrence (argmin)
    }
    idx[b * 16 + m] = bi;
  }
}

// ------------------------------------------------------------------
// Kernel 4: out[b,m] = x[b, idx[b,m]]  (float4 indexed copy, 77 MB)
// ------------------------------------------------------------------
__global__ __launch_bounds__(256) void gather_kernel(
    const float* __restrict__ x, const int* __restrict__ idx,
    float* __restrict__ out)
{
  const int FR4 = 3 * NPIX / 4;            // 37632 float4 per frame
  const float4* x4 = (const float4*)x;
  float4* o4 = (float4*)out;
  size_t g = (size_t)blockIdx.x * 256 + threadIdx.x;
  size_t fo = g / FR4;                     // output frame 0..127
  size_t off = g - fo * FR4;
  int b = (int)(fo >> 4);
  int l = idx[fo];
  o4[g] = x4[((size_t)(b * LLEN) + l) * FR4 + off];
}

extern "C" void kernel_launch(void* const* d_in, const int* in_sizes, int n_in,
                              void* d_out, int out_size, void* d_ws, size_t ws_size,
                              hipStream_t stream) {
  (void)in_sizes; (void)n_in; (void)out_size; (void)ws_size;
  const float* x  = (const float*)d_in[0];
  const float* cw = (const float*)d_in[1];
  const float* cb = (const float*)d_in[2];
  float* out = (float*)d_out;

  // workspace layout
  float* yw    = (float*)d_ws;                       // 512*50176 f32 = 102.8 MB
  float* diffw = yw + (size_t)NFR * NPIX;            // 8*63 f32
  int*   idxw  = (int*)(diffw + 1024);               // 8*16 int (padded)

  // 1) conv as WMMA box filter: 512 frames x 7 row strips
  conv_boxsum_wmma<<<NFR * RBLOCKS, 256, 0, stream>>>(x, cw, cb, yw);

  // 2) pairwise-distance spatial sums
  diff_kernel<<<BATCH * (LLEN - 1), 256, 0, stream>>>(yw, diffw);

  // 3) cumulative-distribution frame selection
  select_kernel<<<1, 64, 0, stream>>>(diffw, idxw);

  // 4) gather selected frames to output
  gather_kernel<<<(BATCH * 16 * (3 * NPIX / 4)) / 256, 256, 0, stream>>>(
      x, idxw, out);
}
